// JointCodebookPredictor_46454366273877
// MI455X (gfx1250) — matrix operand order repaired and checked by reading, not compiled, not run on hardware
//
#include <hip/hip_runtime.h>
#include <stdint.h>

#define PDIM 512
#define NCB  8
#define CS   256
#define HD   256
#define TOTD 2048
#define LN_EPS 1e-5f

typedef __attribute__((ext_vector_type(16))) __bf16 v16bf;
typedef __attribute__((ext_vector_type(8)))  float  v8f;

union ABfrag { v16bf v; uint32_t u[8]; uint4 q[2]; };
union Cfrag  { v8f   v; float   f[8]; };

__device__ __forceinline__ unsigned short f32_to_bf16(float f) {
    union { float f; uint32_t u; } c; c.f = f;
    uint32_t u = c.u;
    uint32_t r = u + 0x7fffu + ((u >> 16) & 1u);   // round-to-nearest-even
    return (unsigned short)(r >> 16);
}

// CDNA5 async copy: 16B global -> LDS, tracked by ASYNCcnt.
// Generic pointers to __shared__ carry the LDS byte offset in their low 32
// bits (LDS aperture mapping truncates to addr[31:0]), so we can feed them
// directly as the VDST LDS-address VGPR.
__device__ __forceinline__ void async_cp16(void* lds_dst, const void* gsrc) {
    uint32_t l = (uint32_t)(uintptr_t)lds_dst;
    asm volatile("global_load_async_to_lds_b128 %0, %1, off"
                 :: "v"(l), "v"(gsrc) : "memory");
}
__device__ __forceinline__ void async_wait0() {
    asm volatile("s_wait_asynccnt 0x0" ::: "memory");
}

#define LOAD_A(frag, kt_) do {                                   \
    const int ka_ = (kt_) * 32 + koffA;                          \
    (frag).q[0] = *(const uint4*)(aBase + ka_);                  \
    (frag).q[1] = *(const uint4*)(aBase + ka_ + 8); } while (0)
#define LOAD_B(frag, kt_) do {                                   \
    const int kb_ = (kt_) * 32 + koffB;                          \
    (frag).q[0] = *(const uint4*)(bBase + kb_);                  \
    (frag).q[1] = *(const uint4*)(bBase + kb_ + 8); } while (0)

__global__ void k_zero(float* out) {
    out[0] = 0.f; out[1] = 0.f;
}

__global__ void k_cvt_bf16(const float* __restrict__ src,
                           unsigned short* __restrict__ dst, int n) {
    for (int i = blockIdx.x * blockDim.x + threadIdx.x; i < n;
         i += gridDim.x * blockDim.x)
        dst[i] = f32_to_bf16(src[i]);
}

// out[c][r] = in[r][c], n x n, 32x32 LDS tiles
__global__ void k_transpose(const float* __restrict__ src,
                            float* __restrict__ dst, int n) {
    __shared__ float tile[32][33];
    const int bx = blockIdx.x * 32, by = blockIdx.y * 32;
    const int tx = threadIdx.x, ty = threadIdx.y;   // 32 x 8
    #pragma unroll
    for (int i = 0; i < 32; i += 8) {
        int r = by + ty + i, c = bx + tx;
        tile[ty + i][tx] = (r < n && c < n) ? src[(size_t)r * n + c] : 0.f;
    }
    __syncthreads();
    #pragma unroll
    for (int i = 0; i < 32; i += 8) {
        int r = bx + ty + i, c = by + tx;
        if (r < n && c < n) dst[(size_t)r * n + c] = tile[tx][ty + i];
    }
}

// ---------------------------------------------------------------------------
// GEMM1 fused: hidden = pred@W1^T + b1 ; hidden[:,CS:] += self gather ;
//              relu ; layernorm ; -> hB (bf16)
// block: 32 rows, 256 threads (8 waves). dyn LDS: 32*512*2 + 32*2048*4 bytes.
// ---------------------------------------------------------------------------
__global__ void __launch_bounds__(256)
k_gemm1(const unsigned short* __restrict__ predB,
        const unsigned short* __restrict__ W1B,
        const float* __restrict__ b1,
        const int*   __restrict__ idx,
        const float* __restrict__ lsT,     // (1792 in) x (1792 out), transposed
        const float* __restrict__ gamma,
        const float* __restrict__ beta,
        unsigned short* __restrict__ hB) {
    extern __shared__ char smem[];
    unsigned short* sA   = (unsigned short*)smem;            // 32 x 512 bf16
    float*          sHid = (float*)(smem + 32 * PDIM * 2);   // 32 x 2048 f32

    const int tid  = threadIdx.x;
    const int lane = tid & 31;
    const int wave = tid >> 5;
    const int row0 = blockIdx.x * 32;

    // async-stage the A tile straight into LDS (ASYNCcnt path)
    {
        const uint4* g = (const uint4*)(predB + (size_t)row0 * PDIM);
        uint4* s = (uint4*)sA;
        for (int i = tid; i < 32 * PDIM / 8; i += 256)
            async_cp16(s + i, g + i);
        async_wait0();
    }
    __syncthreads();

    const int ln15   = lane & 15;
    const int hiHalf = lane >> 4;        // 0 | 1
    const int koffA  = hiHalf * 16;      // A: K sub-offset per lane half
    const int koffB  = hiHalf * 16;      // B: K sub-offset per lane half

    // 2 row-tiles x 128 col-tiles = 256 WMMA tiles, 32 per wave
    for (int t = wave; t < 256; t += 8) {
        const int mt = t & 1;
        const int nt = t >> 1;
        const unsigned short* aBase = sA + (mt * 16 + ln15) * PDIM;
        const unsigned short* bBase = W1B + (size_t)(nt * 16 + ln15) * PDIM;

        // dual accumulators -> consecutive WMMAs independent
        Cfrag c0, c1;
        #pragma unroll
        for (int r = 0; r < 8; ++r) { c0.f[r] = 0.f; c1.f[r] = 0.f; }

        // 4-slot rotating buffers, prefetch distance 2
        ABfrag A[4], B[4];
        LOAD_A(A[0], 0); LOAD_B(B[0], 0);
        LOAD_A(A[1], 1); LOAD_B(B[1], 1);
        #pragma unroll
        for (int kt = 0; kt < PDIM / 32; ++kt) {
            const int cur = kt & 3;
            const int nxt = (kt + 2) & 3;
            if (kt + 2 < PDIM / 32) { LOAD_A(A[nxt], kt + 2); LOAD_B(B[nxt], kt + 2); }
            if (kt & 1)
                c1.v = __builtin_amdgcn_wmma_f32_16x16x32_bf16(
                           false, A[cur].v, false, B[cur].v, (short)0, c1.v, false, false);
            else
                c0.v = __builtin_amdgcn_wmma_f32_16x16x32_bf16(
                           false, A[cur].v, false, B[cur].v, (short)0, c0.v, false, false);
        }
        #pragma unroll
        for (int r = 0; r < 8; ++r) {
            const int m = mt * 16 + hiHalf * 8 + r;
            sHid[m * TOTD + nt * 16 + ln15] = c0.f[r] + c1.f[r];
        }
    }
    __syncthreads();

    // epilogue: 4 rows per wave, 64 cols per lane
    for (int rr = 0; rr < 4; ++rr) {
        const int row = wave * 4 + rr;
        const int rg  = row0 + row;
        int iv[NCB];
        #pragma unroll
        for (int c = 0; c < NCB; ++c) iv[c] = idx[(size_t)rg * NCB + c];

        float s = 0.f, s2 = 0.f;
        for (int i = 0; i < TOTD / 32; ++i) {
            const int col = lane + i * 32;
            float v = sHid[row * TOTD + col] + b1[col];
            if (col >= CS) {                     // self-prediction gather
                const int sr = col - CS;         // 0..1791
                const int j  = sr >> 8;          // output block 0..6
                for (int c = 0; c <= j; ++c) {
                    const int ic = iv[c];
                    if (ic >= 0)
                        v += lsT[(size_t)(c * CS + ic) * (7 * CS) + sr];
                }
            }
            v = fmaxf(v, 0.f);
            sHid[row * TOTD + col] = v;
            s += v; s2 += v * v;
        }
        #pragma unroll
        for (int off = 16; off > 0; off >>= 1) {
            s  += __shfl_xor(s,  off, 32);
            s2 += __shfl_xor(s2, off, 32);
        }
        const float mu  = s * (1.f / TOTD);
        const float var = s2 * (1.f / TOTD) - mu * mu;
        const float rs  = rsqrtf(var + LN_EPS);
        for (int i = 0; i < TOTD / 32; ++i) {
            const int col = lane + i * 32;
            const float v = sHid[row * TOTD + col];
            const float o = (v - mu) * rs * gamma[col] + beta[col];
            hB[(size_t)rg * TOTD + col] = f32_to_bf16(o);
        }
    }
}

// ---------------------------------------------------------------------------
// GEMM2 fused: logits = h@W2^T + b2 ; per-256-chunk log_softmax ; one-hot
//              gather accumulate -> out[0]=tot_logprob, out[1]=tot_count
// block: 32 rows, 256 threads. dyn LDS: 32*2048*2 + 32*256*4 bytes.
// ---------------------------------------------------------------------------
__global__ void __launch_bounds__(256)
k_gemm2(const unsigned short* __restrict__ hBm,
        const unsigned short* __restrict__ W2B,
        const float* __restrict__ b2,
        const int*   __restrict__ idx,
        float* __restrict__ out) {
    extern __shared__ char smem[];
    unsigned short* sH = (unsigned short*)smem;           // 32 x 2048 bf16
    float*          sC = (float*)(smem + 32 * TOTD * 2);  // 32 x 256 f32

    const int tid  = threadIdx.x;
    const int lane = tid & 31;
    const int wave = tid >> 5;
    const int row0 = blockIdx.x * 32;

    // async-stage the h tile straight into LDS (ASYNCcnt path)
    {
        const uint4* g = (const uint4*)(hBm + (size_t)row0 * TOTD);
        uint4* s = (uint4*)sH;
        for (int i = tid; i < 32 * TOTD / 8; i += 256)
            async_cp16(s + i, g + i);
        async_wait0();
    }
    __syncthreads();

    const int ln15   = lane & 15;
    const int hiHalf = lane >> 4;
    const int koffA  = hiHalf * 16;
    const int koffB  = hiHalf * 16;

    float acc = 0.f, cnt = 0.f;

    for (int chunk = 0; chunk < NCB; ++chunk) {
        // 2 row-tiles x 16 col-tiles, 4 per wave
        for (int t = wave; t < 32; t += 8) {
            const int mt  = t & 1;
            const int ntl = t >> 1;                         // 0..15
            const int wrow = chunk * CS + ntl * 16 + ln15;  // W2 row = out col
            const unsigned short* aBase = sH + (mt * 16 + ln15) * TOTD;
            const unsigned short* bBase = W2B + (size_t)wrow * TOTD;

            Cfrag c0, c1;
            #pragma unroll
            for (int r = 0; r < 8; ++r) { c0.f[r] = 0.f; c1.f[r] = 0.f; }

            ABfrag A[4], B[4];
            LOAD_A(A[0], 0); LOAD_B(B[0], 0);
            LOAD_A(A[1], 1); LOAD_B(B[1], 1);
            #pragma unroll 8
            for (int kt = 0; kt < TOTD / 32; ++kt) {
                const int cur = kt & 3;
                const int nxt = (kt + 2) & 3;
                if (kt + 2 < TOTD / 32) { LOAD_A(A[nxt], kt + 2); LOAD_B(B[nxt], kt + 2); }
                if (kt + 4 < TOTD / 32)
                    __builtin_prefetch(bBase + (kt + 4) * 32 + koffB, 0, 1);
                if (kt & 1)
                    c1.v = __builtin_amdgcn_wmma_f32_16x16x32_bf16(
                               false, A[cur].v, false, B[cur].v, (short)0, c1.v, false, false);
                else
                    c0.v = __builtin_amdgcn_wmma_f32_16x16x32_bf16(
                               false, A[cur].v, false, B[cur].v, (short)0, c0.v, false, false);
            }
            const float bias = b2[wrow];
            #pragma unroll
            for (int r = 0; r < 8; ++r) {
                const int m = mt * 16 + hiHalf * 8 + r;
                sC[m * CS + ntl * 16 + ln15] = c0.f[r] + c1.f[r] + bias;
            }
        }
        __syncthreads();

        // log-softmax over this 256-chunk, 4 rows per wave
        for (int rr = 0; rr < 4; ++rr) {
            const int row = wave * 4 + rr;
            const int rg  = row0 + row;
            float x[8];
            float mx = -3.402823466e38f;
            #pragma unroll
            for (int i = 0; i < 8; ++i) {
                x[i] = sC[row * CS + lane + i * 32];
                mx = fmaxf(mx, x[i]);
            }
            #pragma unroll
            for (int off = 16; off > 0; off >>= 1)
                mx = fmaxf(mx, __shfl_xor(mx, off, 32));
            float se = 0.f;
            #pragma unroll
            for (int i = 0; i < 8; ++i) se += __expf(x[i] - mx);
            #pragma unroll
            for (int off = 16; off > 0; off >>= 1)
                se += __shfl_xor(se, off, 32);
            const int ic = idx[(size_t)rg * NCB + chunk];
            if (ic >= 0) {
                acc += sC[row * CS + ic] - mx - __logf(se);
                cnt += 1.f;
            }
        }
        __syncthreads();
    }
    if (lane == 0) {
        atomicAdd(&out[0], acc);
        atomicAdd(&out[1], cnt);
    }
}

extern "C" void kernel_launch(void* const* d_in, const int* in_sizes, int n_in,
                              void* d_out, int out_size, void* d_ws, size_t ws_size,
                              hipStream_t stream) {
    const float* pred  = (const float*)d_in[0];   // (16,2048,512)
    const int*   idx   = (const int*)  d_in[1];   // (16,2048,8)
    const float* W1    = (const float*)d_in[2];   // (2048,512)
    const float* b1    = (const float*)d_in[3];   // (2048,)
    const float* lself = (const float*)d_in[4];   // (1792,1792)
    const float* gamma = (const float*)d_in[5];   // (2048,)
    const float* beta  = (const float*)d_in[6];   // (2048,)
    const float* W2    = (const float*)d_in[7];   // (2048,2048)
    const float* b2    = (const float*)d_in[8];   // (2048,)
    float* out = (float*)d_out;

    char* ws = (char*)d_ws;
    unsigned short* predB = (unsigned short*)(ws);              // 32 MB
    unsigned short* W1B   = (unsigned short*)(ws + 33554432);   //  2 MB
    unsigned short* W2B   = (unsigned short*)(ws + 35651584);   //  8 MB
    float*          lsT   = (float*)         (ws + 44040192);   // 12.8 MB
    unsigned short* hB    = (unsigned short*)(ws + 56885248);   // 128 MB

    k_zero<<<1, 1, 0, stream>>>(out);
    k_cvt_bf16<<<2048, 256, 0, stream>>>(pred, predB, 32768 * PDIM);
    k_cvt_bf16<<<512,  256, 0, stream>>>(W1, W1B, TOTD * PDIM);
    k_cvt_bf16<<<2048, 256, 0, stream>>>(W2, W2B, TOTD * TOTD);
    k_transpose<<<dim3(56, 56), dim3(32, 8), 0, stream>>>(lself, lsT, 7 * CS);

    const size_t smem1 = 32 * PDIM * 2 + 32 * TOTD * 4;  // 288 KB
    k_gemm1<<<1024, 256, smem1, stream>>>(predB, W1B, b1, idx, lsT,
                                          gamma, beta, hB);
    const size_t smem2 = 32 * TOTD * 2 + 32 * CS * 4;    // 160 KB
    k_gemm2<<<1024, 256, smem2, stream>>>(hB, W2B, b2, idx, out);
}
